// CurvSelfAttention_81784767250783
// MI455X (gfx1250) — matrix operand order repaired and checked by reading, not compiled, not run on hardware
//
#include <hip/hip_runtime.h>
#include <hip/hip_bf16.h>

// ---------------------------------------------------------------------------
// Problem constants
// ---------------------------------------------------------------------------
#define Bn   2
#define Sn   2048
#define En   1024
#define Hn   16
#define Dn   64
#define Gn   16
#define Nall 3328          // 3*E + H*G
#define Mall 4096          // B*S

typedef _Float16 v4h  __attribute__((ext_vector_type(4)));
typedef _Float16 v8h  __attribute__((ext_vector_type(8)));
typedef _Float16 v16h __attribute__((ext_vector_type(16)));
typedef float    v8f  __attribute__((ext_vector_type(8)));

union ABFrag { v16h v; v8h h[2]; };

// A-operand (16-bit 16x32): per lane one contiguous 16-halfword chunk at
// K = (lane>>4)*16.  `p` must already point at (row base + kh).
__device__ __forceinline__ v16h load_fragA(const _Float16* p) {
    ABFrag f;
    f.h[0] = *(const v8h*)(p);
    f.h[1] = *(const v8h*)(p + 8);
    return f.v;
}
// B-operand (16-bit 32x16): per lane one contiguous 16-halfword chunk at
// K = (lane>>4)*16.  `p` must already point at (col base + kh).
__device__ __forceinline__ v16h load_fragB(const _Float16* p) {
    ABFrag f;
    f.h[0] = *(const v8h*)(p);
    f.h[1] = *(const v8h*)(p + 8);
    return f.v;
}

// ---------------------------------------------------------------------------
// Kernel 0a: pack Wq|Wk|Wv|Ws -> f16 [3328][1024], and biases -> f32 [3328]
// ---------------------------------------------------------------------------
__global__ void pack_weights(const float* __restrict__ Wq, const float* __restrict__ Wk,
                             const float* __restrict__ Wv, const float* __restrict__ Ws,
                             const float* __restrict__ bq, const float* __restrict__ bk,
                             const float* __restrict__ bv, const float* __restrict__ bs,
                             _Float16* __restrict__ Wf, float* __restrict__ biasf) {
    size_t t = (size_t)blockIdx.x * blockDim.x + threadIdx.x;
    if (t >= (size_t)Nall * En) return;
    int n = (int)(t >> 10);
    int k = (int)(t & 1023);
    float v;
    if (n < 1024)       v = Wq[(size_t)n * En + k];
    else if (n < 2048)  v = Wk[(size_t)(n - 1024) * En + k];
    else if (n < 3072)  v = Wv[(size_t)(n - 2048) * En + k];
    else                v = Ws[(size_t)(n - 3072) * En + k];
    Wf[t] = (_Float16)v;
    if (k == 0) {
        float b;
        if (n < 1024)       b = bq[n];
        else if (n < 2048)  b = bk[n - 1024];
        else if (n < 3072)  b = bv[n - 2048];
        else                b = bs[n - 3072];
        biasf[n] = b;
    }
}

// ---------------------------------------------------------------------------
// Kernel 0b: hidden_states f32 -> f16 [4096][1024]
// ---------------------------------------------------------------------------
__global__ void conv_x(const float* __restrict__ X, _Float16* __restrict__ Xf) {
    size_t t = (size_t)blockIdx.x * blockDim.x + threadIdx.x;
    if (t >= (size_t)Mall * En) return;
    Xf[t] = (_Float16)X[t];
}

// ---------------------------------------------------------------------------
// Kernel 1: fused QKVS projection GEMM.  C[m,n] = sum_k Xf[m,k]*Wf[n,k] + bias.
// Each wave: 32 (M) x 64 (N) tile, k-step 32 -> 8x v_wmma_f32_16x16x32_f16,
// with explicit next-step fragment prefetch (double-buffered registers).
// Epilogue scatters: Q,K -> [B,H,S,D] f16 ; V -> [B,H,D,S] f16 ; slog -> f32.
// ---------------------------------------------------------------------------
__global__ __launch_bounds__(256) void qkvs_gemm(
    const _Float16* __restrict__ Xf, const _Float16* __restrict__ Wf,
    const float* __restrict__ biasf,
    _Float16* __restrict__ Qf, _Float16* __restrict__ Kf,
    _Float16* __restrict__ Vt, float* __restrict__ Slog) {

    const int lane  = threadIdx.x & 31;
    const int wave  = threadIdx.x >> 5;
    const int row   = lane & 15;
    const int half8 = (lane >> 4) << 3;   // C-layout half offset
    const int kh    = (lane >> 4) << 4;   // B-frag half offset

    const int m0 = blockIdx.y * 256 + wave * 32;   // two 16-row M subtiles
    const int n0 = blockIdx.x * 64;

    const _Float16* aP0 = Xf + (size_t)(m0 + row) * En + kh;
    const _Float16* aP1 = aP0 + (size_t)16 * En;
    const _Float16* bP  = Wf + (size_t)(n0 + row) * En + kh;

    v8f zero = {};
    v8f acc[2][4] = {{zero, zero, zero, zero}, {zero, zero, zero, zero}};

    // prologue: fragments for k0 = 0
    v16h a0 = load_fragA(aP0);
    v16h a1 = load_fragA(aP1);
    v16h b[4];
#pragma unroll
    for (int j = 0; j < 4; ++j) b[j] = load_fragB(bP + (size_t)j * 16 * En);

#pragma unroll 2
    for (int k0 = 0; k0 < En; k0 += 32) {
        const int k1 = (k0 + 32) & (En - 1);   // wrapped, branch-free prefetch
        // prefetch next-step fragments into the alternate register set
        v16h na0 = load_fragA(aP0 + k1);
        v16h na1 = load_fragA(aP1 + k1);
        v16h nb[4];
#pragma unroll
        for (int j = 0; j < 4; ++j) nb[j] = load_fragB(bP + (size_t)j * 16 * En + k1);

        // 8 WMMAs on the current fragments (overlap with prefetch loads)
#pragma unroll
        for (int j = 0; j < 4; ++j)
            acc[0][j] = __builtin_amdgcn_wmma_f32_16x16x32_f16(
                false, a0, false, b[j], (short)0, acc[0][j], false, false);
#pragma unroll
        for (int j = 0; j < 4; ++j)
            acc[1][j] = __builtin_amdgcn_wmma_f32_16x16x32_f16(
                false, a1, false, b[j], (short)0, acc[1][j], false, false);

        a0 = na0; a1 = na1;
#pragma unroll
        for (int j = 0; j < 4; ++j) b[j] = nb[j];
    }

    const int bidx = m0 >> 11;             // batch (2048 rows each)

#pragma unroll
    for (int mi = 0; mi < 2; ++mi) {
        const int srow0 = ((m0 + mi * 16) & 2047) + half8;
#pragma unroll
        for (int j = 0; j < 4; ++j) {
            const int n  = n0 + j * 16 + row;
            const float bv = biasf[n];
            if (n < 2048) {                       // Q or K -> [B,H,S,D] f16
                _Float16* dst = (n < 1024) ? Qf : Kf;
                const int nn = n & 1023;
                const int h = nn >> 6, d = nn & 63;
                const size_t base = ((size_t)(bidx * Hn + h)) * Sn * Dn + d;
#pragma unroll
                for (int r = 0; r < 8; ++r)
                    dst[base + (size_t)(srow0 + r) * Dn] =
                        (_Float16)(acc[mi][j][r] + bv);
            } else if (n < 3072) {                // V -> [B,H,D,S] f16 (transposed)
                const int nn = n - 2048;
                const int h = nn >> 6, d = nn & 63;
                v8h pk;
#pragma unroll
                for (int r = 0; r < 8; ++r) pk[r] = (_Float16)(acc[mi][j][r] + bv);
                *(v8h*)(Vt + (((size_t)(bidx * Hn + h)) * Dn + d) * Sn + srow0) = pk;
            } else {                              // s logits -> [B,S,256] f32
                const int nn = n - 3072;
#pragma unroll
                for (int r = 0; r < 8; ++r)
                    Slog[((size_t)(bidx * Sn + srow0 + r)) * 256 + nn] =
                        acc[mi][j][r] + bv;
            }
        }
    }
}

// ---------------------------------------------------------------------------
// Kernel 2: Q *= (0.95 + 0.1*sigmoid(slog)) * (1/sqrt(D)) * log2(e)
// One thread per (b,s,h,g) group of 4 contiguous d values.
// ---------------------------------------------------------------------------
__global__ void scale_q(const float* __restrict__ Slog, _Float16* __restrict__ Qf) {
    int t = blockIdx.x * blockDim.x + threadIdx.x;   // B*S*H*G = 1<<20
    int g = t & 15;
    int h = (t >> 4) & 15;
    int s = (t >> 8) & 2047;
    int b = t >> 19;
    float lg  = Slog[((size_t)(b * Sn + s)) * 256 + h * 16 + g];
    float sig = 1.0f / (1.0f + __builtin_expf(-lg));
    float f   = (0.95f + 0.1f * sig) * 0.125f * 1.4426950408889634f;
    v4h* p = (v4h*)(Qf + ((((size_t)(b * Hn + h)) * Sn + s) * Dn + g * 4));
    v4h q = *p;
#pragma unroll
    for (int i = 0; i < 4; ++i) q[i] = (_Float16)((float)q[i] * f);
    *p = q;
}

// ---------------------------------------------------------------------------
// Kernel 3: flash attention.  One wave = 16 query rows; t-loop step 32.
// K fragments for the next step are prefetched while the current step's
// scores/softmax/PV run; V fragments load at the top of the iteration so
// their latency hides under the softmax (shuffles + v_exp_f32).
// ---------------------------------------------------------------------------
__global__ __launch_bounds__(256) void attn(
    const _Float16* __restrict__ Qf, const _Float16* __restrict__ Kf,
    const _Float16* __restrict__ Vt, float* __restrict__ out) {

    __shared__ _Float16 lds[8][16][32];   // per-wave 16x32 P tile

    const int lane  = threadIdx.x & 31;
    const int wave  = threadIdx.x >> 5;
    const int row   = lane & 15;
    const int half8 = (lane >> 4) << 3;
    const int kh    = (lane >> 4) << 4;

    const int bh = blockIdx.x;            // 0..31
    const int b  = bh >> 4, h = bh & 15;
    const int s0 = blockIdx.y * 128 + wave * 16;

    const _Float16* Qb = Qf + (size_t)bh * Sn * Dn;
    const _Float16* Kb = Kf + (size_t)bh * Sn * Dn;
    const _Float16* Vb = Vt + (size_t)bh * Dn * Sn;

    // Q A-fragments (rows s0..s0+15, d split 0..31 / 32..63)
    const _Float16* qp = Qb + (size_t)(s0 + row) * Dn + kh;
    v16h aq0 = load_fragA(qp);
    v16h aq1 = load_fragA(qp + 32);

    float mrow[8], lrow[8];
    v8f zero = {};
    v8f o[4] = {zero, zero, zero, zero};
#pragma unroll
    for (int r = 0; r < 8; ++r) { mrow[r] = -1e30f; lrow[r] = 0.0f; }

    // prologue: K fragments for t0 = 0   ([ti*2+dh]: ti = col tile, dh = d half)
    v16h bk[4];
    {
        const _Float16* kp0 = Kb + (size_t)row * Dn + kh;
        const _Float16* kp1 = Kb + (size_t)(16 + row) * Dn + kh;
        bk[0] = load_fragB(kp0); bk[1] = load_fragB(kp0 + 32);
        bk[2] = load_fragB(kp1); bk[3] = load_fragB(kp1 + 32);
    }

#pragma unroll 2
    for (int t0 = 0; t0 < Sn; t0 += 32) {
        // ---- V fragments for this step: issue early, consumed last ----
        v16h bv[4];
#pragma unroll
        for (int j = 0; j < 4; ++j)
            bv[j] = load_fragB(Vb + (size_t)(j * 16 + row) * Sn + t0 + kh);

        // ---- scores: two 16x16 tiles over columns t0..t0+31 ----
        v8f sc[2];
#pragma unroll
        for (int ti = 0; ti < 2; ++ti) {
            v8f z = {};
            z = __builtin_amdgcn_wmma_f32_16x16x32_f16(false, aq0, false, bk[ti * 2],
                                                       (short)0, z, false, false);
            z = __builtin_amdgcn_wmma_f32_16x16x32_f16(false, aq1, false, bk[ti * 2 + 1],
                                                       (short)0, z, false, false);
            sc[ti] = z;
        }

        // ---- prefetch next step's K fragments (wrapped, branch-free) ----
        const int t1 = (t0 + 32) & (Sn - 1);
        v16h nbk[4];
        {
            const _Float16* kp0 = Kb + (size_t)(t1 + row) * Dn + kh;
            const _Float16* kp1 = Kb + (size_t)(t1 + 16 + row) * Dn + kh;
            nbk[0] = load_fragB(kp0); nbk[1] = load_fragB(kp0 + 32);
            nbk[2] = load_fragB(kp1); nbk[3] = load_fragB(kp1 + 32);
        }

        // ---- online softmax (row stats: xor-shuffle within 16-lane halves) ----
        float corr[8];
#pragma unroll
        for (int r = 0; r < 8; ++r) {
            float v = fmaxf(sc[0][r], sc[1][r]);
            v = fmaxf(v, __shfl_xor(v, 1, 32));
            v = fmaxf(v, __shfl_xor(v, 2, 32));
            v = fmaxf(v, __shfl_xor(v, 4, 32));
            v = fmaxf(v, __shfl_xor(v, 8, 32));
            float mnew = fmaxf(mrow[r], v);
            corr[r] = __builtin_amdgcn_exp2f(mrow[r] - mnew);
            mrow[r] = mnew;
        }
#pragma unroll
        for (int r = 0; r < 8; ++r) {
            float p0 = __builtin_amdgcn_exp2f(sc[0][r] - mrow[r]);
            float p1 = __builtin_amdgcn_exp2f(sc[1][r] - mrow[r]);
            float rs = p0 + p1;
            rs += __shfl_xor(rs, 1, 32);
            rs += __shfl_xor(rs, 2, 32);
            rs += __shfl_xor(rs, 4, 32);
            rs += __shfl_xor(rs, 8, 32);
            lrow[r] = lrow[r] * corr[r] + rs;
            lds[wave][half8 + r][row]      = (_Float16)p0;   // C-layout -> LDS
            lds[wave][half8 + r][row + 16] = (_Float16)p1;
        }
        asm volatile("s_wait_dscnt 0" ::: "memory");  // per-wave DS in-order; fence

        // ---- P as A-fragment, accumulate P x V ----
        v16h ap = load_fragA(&lds[wave][row][kh]);
#pragma unroll
        for (int j = 0; j < 4; ++j) {
#pragma unroll
            for (int r = 0; r < 8; ++r) o[j][r] *= corr[r];
            o[j] = __builtin_amdgcn_wmma_f32_16x16x32_f16(false, ap, false, bv[j],
                                                          (short)0, o[j], false, false);
        }

#pragma unroll
        for (int i = 0; i < 4; ++i) bk[i] = nbk[i];
    }

    // ---- finalize: divide by l, write ctx [B,S,H*D] f32 ----
    float inv[8];
#pragma unroll
    for (int r = 0; r < 8; ++r) inv[r] = 1.0f / lrow[r];
    const size_t obase = ((size_t)b * Sn) * (Hn * Dn) + h * Dn;
#pragma unroll
    for (int j = 0; j < 4; ++j) {
        const int d = j * 16 + row;
#pragma unroll
        for (int r = 0; r < 8; ++r) {
            const int s = s0 + half8 + r;
            out[obase + (size_t)s * (Hn * Dn) + d] = o[j][r] * inv[r];
        }
    }
}

// ---------------------------------------------------------------------------
// Workspace layout (bytes)
// ---------------------------------------------------------------------------
#define OFF_XF    ((size_t)0)                    //  8,388,608  Xf16 [4096][1024]
#define OFF_WF    ((size_t)8388608)              //  6,815,744  Wf16 [3328][1024]
#define OFF_BIAS  ((size_t)15204352)             //     13,312  bias f32
#define OFF_QF    ((size_t)15217664)             //  8,388,608  Q f16 [B,H,S,D]
#define OFF_KF    ((size_t)23606272)             //  8,388,608  K f16 [B,H,S,D]
#define OFF_VT    ((size_t)31994880)             //  8,388,608  V f16 [B,H,D,S]
#define OFF_SLOG  ((size_t)40383488)             //  4,194,304  slog f32 [B,S,256]

extern "C" void kernel_launch(void* const* d_in, const int* in_sizes, int n_in,
                              void* d_out, int out_size, void* d_ws, size_t ws_size,
                              hipStream_t stream) {
    const float* X  = (const float*)d_in[0];
    const float* Wq = (const float*)d_in[1];
    const float* bq = (const float*)d_in[2];
    const float* Wk = (const float*)d_in[3];
    const float* bk = (const float*)d_in[4];
    const float* Wv = (const float*)d_in[5];
    const float* bv = (const float*)d_in[6];
    const float* Ws = (const float*)d_in[7];
    const float* bs = (const float*)d_in[8];

    char* ws = (char*)d_ws;
    _Float16* Xf   = (_Float16*)(ws + OFF_XF);
    _Float16* Wf   = (_Float16*)(ws + OFF_WF);
    float*    bias = (float*)   (ws + OFF_BIAS);
    _Float16* Qf   = (_Float16*)(ws + OFF_QF);
    _Float16* Kf   = (_Float16*)(ws + OFF_KF);
    _Float16* Vt   = (_Float16*)(ws + OFF_VT);
    float*    Slog = (float*)   (ws + OFF_SLOG);
    float*    out  = (float*)d_out;

    // 0a: pack weights + bias (3328*1024 elements)
    pack_weights<<<dim3((Nall * En) / 256), dim3(256), 0, stream>>>(
        Wq, Wk, Wv, Ws, bq, bk, bv, bs, Wf, bias);

    // 0b: convert hidden states (4096*1024 elements)
    conv_x<<<dim3((Mall * En) / 256), dim3(256), 0, stream>>>(X, Xf);

    // 1: fused QKVS projection GEMM (M=4096, N=3328, K=1024)
    qkvs_gemm<<<dim3(Nall / 64, Mall / 256), dim3(256), 0, stream>>>(
        Xf, Wf, bias, Qf, Kf, Vt, Slog);

    // 2: apply dynamic group scale (+ 1/sqrt(D) and log2e fold) to Q
    scale_q<<<dim3((Bn * Sn * Hn * Gn) / 256), dim3(256), 0, stream>>>(Slog, Qf);

    // 3: flash attention; grid = (B*H, S/128), 8 waves x 16 query rows each
    attn<<<dim3(Bn * Hn, Sn / 128), dim3(256), 0, stream>>>(Qf, Kf, Vt, out);
}
